// GATLayerOut_single_1932735283946
// MI455X (gfx1250) — matrix operand order repaired and checked by reading, not compiled, not run on hardware
//
#include <hip/hip_runtime.h>
#include <hip/hip_bf16.h>
#include <math.h>

typedef __attribute__((ext_vector_type(16))) __bf16 v16bf;
typedef __attribute__((ext_vector_type(8)))  float  v8f;
typedef int v4i __attribute__((vector_size(16)));
typedef __attribute__((address_space(1))) v4i* gv4i_t;   // global int4*
typedef __attribute__((address_space(3))) v4i* lv4i_t;   // LDS int4*

#define N_NODES 8192
#define F_IN    256
#define F_OUT   128
#define NEGV    (-9000000000000000.0f)

#if defined(__gfx1250__) && __has_builtin(__builtin_amdgcn_global_load_async_to_lds_b128)
#define USE_ASYNC_LDS 1
#else
#define USE_ASYNC_LDS 0
#endif

// ---------------------------------------------------------------------------
// Pack W (256x128 f32, row-major) into the wave32 WMMA B-operand layout, bf16.
// B layout (16x16x32 bf16): per 32x16 (KxN) tile, lane = (n%16) + 16*((k%16)/8),
// element = (k%8) + 8*((k%32)/16). Packed tile stride = 512 elements. Tile order: kt-major, nt.
// ---------------------------------------------------------------------------
__global__ void pack_W_kernel(const float* __restrict__ W, __bf16* __restrict__ Wp) {
    int t    = blockIdx.x * blockDim.x + threadIdx.x;   // 0..32767 (packed index)
    int elem = t & 15;
    int lane = (t >> 4) & 31;
    int tile = t >> 9;                                  // 0..63
    int kt = tile >> 3, nt = tile & 7;
    int k = kt * 32 + ((elem >> 3) << 4) + ((lane >> 4) << 3) + (elem & 7);
    int n = nt * 16 + (lane & 15);
    Wp[t] = (__bf16)W[(size_t)k * F_OUT + n];
}

// ---------------------------------------------------------------------------
// GEMM1: h = x @ W, one wave per 16x128 output tile.
// Also emits h packed into bf16 B-layout for GEMM2 (h is GEMM2's B operand).
// ---------------------------------------------------------------------------
__global__ void gemm1_kernel(const float* __restrict__ x, const __bf16* __restrict__ Wp,
                             float* __restrict__ h, __bf16* __restrict__ hp) {
    const int lane = threadIdx.x & 31;
    const int tile = blockIdx.x * (blockDim.x >> 5) + (threadIdx.x >> 5); // 0..511
    const int lm = lane & 15, hl = lane >> 4;
    const int i0 = tile * 16;
    const float* xr = x + (size_t)(i0 + lm) * F_IN;

    v8f acc[8] = {};

    for (int k0 = 0; k0 < F_IN; k0 += 32) {
        const float4 f0 = *(const float4*)(xr + k0 + hl * 8);
        const float4 f1 = *(const float4*)(xr + k0 + hl * 8 + 4);
        const float4 f2 = *(const float4*)(xr + k0 + 16 + hl * 8);
        const float4 f3 = *(const float4*)(xr + k0 + 16 + hl * 8 + 4);
        v16bf A;
        A[0]=(__bf16)f0.x;  A[1]=(__bf16)f0.y;  A[2]=(__bf16)f0.z;  A[3]=(__bf16)f0.w;
        A[4]=(__bf16)f1.x;  A[5]=(__bf16)f1.y;  A[6]=(__bf16)f1.z;  A[7]=(__bf16)f1.w;
        A[8]=(__bf16)f2.x;  A[9]=(__bf16)f2.y;  A[10]=(__bf16)f2.z; A[11]=(__bf16)f2.w;
        A[12]=(__bf16)f3.x; A[13]=(__bf16)f3.y; A[14]=(__bf16)f3.z; A[15]=(__bf16)f3.w;

        const __bf16* bp = Wp + (size_t)(k0 >> 5) * 8 * 512;
        #pragma unroll
        for (int nt = 0; nt < 8; ++nt) {
            v16bf B = *(const v16bf*)(bp + (size_t)nt * 512 + lane * 16);
            acc[nt] = __builtin_amdgcn_wmma_f32_16x16x32_bf16(
                false, A, false, B, (short)0, acc[nt], false, false);
        }
    }

    const int kb   = (i0 >> 5) * 8;          // hp k-tile base
    const int lb16 = 16 * ((i0 >> 4) & 1);   // k-half inside hp tile (element offset lb16/2)
    #pragma unroll
    for (int nt = 0; nt < 8; ++nt) {
        #pragma unroll
        for (int r = 0; r < 8; ++r) {
            int orow = i0 + r + 8 * hl;
            int ocol = nt * 16 + lm;
            float v = acc[nt][r];
            h[(size_t)orow * F_OUT + ocol] = v;
            hp[(size_t)(kb + nt) * 512 + (lm + 16 * hl) * 16 + (r + (lb16 >> 1))] = (__bf16)v;
        }
    }
}

// ---------------------------------------------------------------------------
// s1 = h@a1, s2 = h@a2. One wave32 per row.
// ---------------------------------------------------------------------------
__global__ void scores_kernel(const float* __restrict__ h, const float* __restrict__ a,
                              float* __restrict__ s1, float* __restrict__ s2) {
    int row  = (blockIdx.x * blockDim.x + threadIdx.x) >> 5;
    int lane = threadIdx.x & 31;
    if (row >= N_NODES) return;
    const float* hr = h + (size_t)row * F_OUT;
    float p1 = 0.f, p2 = 0.f;
    #pragma unroll
    for (int c = 0; c < F_OUT; c += 32) {
        float hv = hr[c + lane];
        p1 = fmaf(hv, a[c + lane], p1);
        p2 = fmaf(hv, a[F_OUT + c + lane], p2);
    }
    #pragma unroll
    for (int off = 16; off > 0; off >>= 1) {
        p1 += __shfl_down(p1, off);
        p2 += __shfl_down(p2, off);
    }
    if (lane == 0) { s1[row] = p1; s2[row] = p2; }
}

// ---------------------------------------------------------------------------
// Online masked-softmax statistics per row: rowmax, 1/rowsum. One wave per row.
// adj pass #1 (HBM-bound, 268 MB).
// ---------------------------------------------------------------------------
__global__ void softmax_stats_kernel(const int* __restrict__ adj,
                                     const float* __restrict__ s1,
                                     const float* __restrict__ s2,
                                     float* __restrict__ rmax, float* __restrict__ rinv) {
    int row  = (blockIdx.x * blockDim.x + threadIdx.x) >> 5;
    int lane = threadIdx.x & 31;
    if (row >= N_NODES) return;
    const int* ar = adj + (size_t)row * N_NODES;
    const float s1v = s1[row];
    float m = -INFINITY, s = 0.f;
    for (int j = lane; j < N_NODES; j += 32) {
        float e = s1v + s2[j];
        e = (e >= 0.f) ? e : 0.01f * e;          // leaky_relu(0.01)
        float v = (ar[j] > 0) ? e : NEGV;
        float mn = fmaxf(m, v);
        s = s * expf(m - mn) + expf(v - mn);
        m = mn;
    }
    #pragma unroll
    for (int off = 16; off > 0; off >>= 1) {
        float m2 = __shfl_xor(m, off);
        float sx = __shfl_xor(s, off);
        float mn = fmaxf(m, m2);
        s = s * expf(m - mn) + sx * expf(m2 - mn);
        m = mn;
    }
    if (lane == 0) { rmax[row] = m; rinv[row] = 1.0f / s; }
}

// ---------------------------------------------------------------------------
// GEMM2 + elu: out = elu(attention @ h).
// Block = 4 waves; each wave owns two 16-row tiles (M=32/wave). B K-step tiles
// (8 KB of packed h) are staged once per block into double-buffered LDS via the
// gfx1250 async global->LDS path (fallback: ds_store staging), then consumed by
// all 4 waves: 8x less L2 B-traffic than the register-direct version.
// ---------------------------------------------------------------------------
__device__ __forceinline__ void stage_tile(__bf16* dst, const __bf16* src, int t) {
#if USE_ASYNC_LDS
    #pragma unroll
    for (int c = 0; c < 4; ++c) {
        __builtin_amdgcn_global_load_async_to_lds_b128(
            (gv4i_t)(src + c * 1024 + t * 8),
            (lv4i_t)(dst + c * 1024 + t * 8),
            0, 0);
    }
#else
    #pragma unroll
    for (int c = 0; c < 4; ++c) {
        *(int4*)(dst + c * 1024 + t * 8) = *(const int4*)(src + c * 1024 + t * 8);
    }
#endif
}

__device__ __forceinline__ void wait_async_le(int pending) {
#if USE_ASYNC_LDS
    if (pending) asm volatile("s_wait_asynccnt 0x4" ::: "memory");
    else         asm volatile("s_wait_asynccnt 0x0" ::: "memory");
#endif
}

__device__ __forceinline__ __bf16 attv(int flag, float s2v, float s1v,
                                       float mv, float inv, float zatt) {
    float e = s1v + s2v;
    e = (e >= 0.f) ? e : 0.01f * e;
    return (__bf16)((flag > 0) ? expf(e - mv) * inv : zatt);
}

__global__ void __launch_bounds__(128)
gemm2_elu_kernel(const int* __restrict__ adj,
                 const float* __restrict__ s1, const float* __restrict__ s2,
                 const float* __restrict__ rmax, const float* __restrict__ rinv,
                 const __bf16* __restrict__ hp, float* __restrict__ out) {
    __shared__ __align__(32) __bf16 sB[2][4096];   // 2 x 8 KB K-step B tiles

    const int t    = threadIdx.x;
    const int lane = t & 31;
    const int wv   = t >> 5;                               // 0..3
    const int lm = lane & 15, hl = lane >> 4;
    const int tb  = (blockIdx.x * 4 + wv) * 2;             // this wave's tile pair
    const int i0a = tb * 16, i0b = i0a + 16;
    const int rowA = i0a + lm, rowB = i0b + lm;

    const float s1vA = s1[rowA], mvA = rmax[rowA], invA = rinv[rowA];
    const float s1vB = s1[rowB], mvB = rmax[rowB], invB = rinv[rowB];
    const float zattA = expf(NEGV - mvA) * invA;   // 0 normally; 1/8192 all-masked
    const float zattB = expf(NEGV - mvB) * invB;
    const int* arA = adj + (size_t)rowA * N_NODES;
    const int* arB = adj + (size_t)rowB * N_NODES;

    v8f accA[8] = {};
    v8f accB[8] = {};

    stage_tile(&sB[0][0], hp, t);

    for (int jt = 0; jt < N_NODES / 32; ++jt) {
        const int cur = jt & 1;
        if (jt + 1 < N_NODES / 32) {
            stage_tile(&sB[cur ^ 1][0], hp + (size_t)(jt + 1) * 4096, t);
            wait_async_le(1);
        } else {
            wait_async_le(0);
        }
        __syncthreads();

        const int j0 = jt * 32;
        if (j0 + 1024 < N_NODES) {
            __builtin_prefetch((const void*)(arA + j0 + 1024), 0, 1);
            __builtin_prefetch((const void*)(arB + j0 + 1024), 0, 1);
        }

        const int jA = j0 + hl * 8;
        const int jB = j0 + 16 + hl * 8;
        const int4   m0 = *(const int4*)(arA + jA);
        const int4   m1 = *(const int4*)(arA + jA + 4);
        const int4   m2 = *(const int4*)(arA + jB);
        const int4   m3 = *(const int4*)(arA + jB + 4);
        const int4   n0 = *(const int4*)(arB + jA);
        const int4   n1 = *(const int4*)(arB + jA + 4);
        const int4   n2 = *(const int4*)(arB + jB);
        const int4   n3 = *(const int4*)(arB + jB + 4);
        const float4 t0 = *(const float4*)(s2 + jA);
        const float4 t1 = *(const float4*)(s2 + jA + 4);
        const float4 t2 = *(const float4*)(s2 + jB);
        const float4 t3 = *(const float4*)(s2 + jB + 4);

        v16bf A0, A1;
        A0[0]  = attv(m0.x, t0.x, s1vA, mvA, invA, zattA);
        A0[1]  = attv(m0.y, t0.y, s1vA, mvA, invA, zattA);
        A0[2]  = attv(m0.z, t0.z, s1vA, mvA, invA, zattA);
        A0[3]  = attv(m0.w, t0.w, s1vA, mvA, invA, zattA);
        A0[4]  = attv(m1.x, t1.x, s1vA, mvA, invA, zattA);
        A0[5]  = attv(m1.y, t1.y, s1vA, mvA, invA, zattA);
        A0[6]  = attv(m1.z, t1.z, s1vA, mvA, invA, zattA);
        A0[7]  = attv(m1.w, t1.w, s1vA, mvA, invA, zattA);
        A0[8]  = attv(m2.x, t2.x, s1vA, mvA, invA, zattA);
        A0[9]  = attv(m2.y, t2.y, s1vA, mvA, invA, zattA);
        A0[10] = attv(m2.z, t2.z, s1vA, mvA, invA, zattA);
        A0[11] = attv(m2.w, t2.w, s1vA, mvA, invA, zattA);
        A0[12] = attv(m3.x, t3.x, s1vA, mvA, invA, zattA);
        A0[13] = attv(m3.y, t3.y, s1vA, mvA, invA, zattA);
        A0[14] = attv(m3.z, t3.z, s1vA, mvA, invA, zattA);
        A0[15] = attv(m3.w, t3.w, s1vA, mvA, invA, zattA);

        A1[0]  = attv(n0.x, t0.x, s1vB, mvB, invB, zattB);
        A1[1]  = attv(n0.y, t0.y, s1vB, mvB, invB, zattB);
        A1[2]  = attv(n0.z, t0.z, s1vB, mvB, invB, zattB);
        A1[3]  = attv(n0.w, t0.w, s1vB, mvB, invB, zattB);
        A1[4]  = attv(n1.x, t1.x, s1vB, mvB, invB, zattB);
        A1[5]  = attv(n1.y, t1.y, s1vB, mvB, invB, zattB);
        A1[6]  = attv(n1.z, t1.z, s1vB, mvB, invB, zattB);
        A1[7]  = attv(n1.w, t1.w, s1vB, mvB, invB, zattB);
        A1[8]  = attv(n2.x, t2.x, s1vB, mvB, invB, zattB);
        A1[9]  = attv(n2.y, t2.y, s1vB, mvB, invB, zattB);
        A1[10] = attv(n2.z, t2.z, s1vB, mvB, invB, zattB);
        A1[11] = attv(n2.w, t2.w, s1vB, mvB, invB, zattB);
        A1[12] = attv(n3.x, t3.x, s1vB, mvB, invB, zattB);
        A1[13] = attv(n3.y, t3.y, s1vB, mvB, invB, zattB);
        A1[14] = attv(n3.z, t3.z, s1vB, mvB, invB, zattB);
        A1[15] = attv(n3.w, t3.w, s1vB, mvB, invB, zattB);

        #pragma unroll
        for (int nt = 0; nt < 8; ++nt) {
            v16bf B = *(const v16bf*)(&sB[cur][nt * 512 + lane * 16]);
            accA[nt] = __builtin_amdgcn_wmma_f32_16x16x32_bf16(
                false, A0, false, B, (short)0, accA[nt], false, false);
            accB[nt] = __builtin_amdgcn_wmma_f32_16x16x32_bf16(
                false, A1, false, B, (short)0, accB[nt], false, false);
        }
        __syncthreads();
    }

    #pragma unroll
    for (int nt = 0; nt < 8; ++nt) {
        #pragma unroll
        for (int r = 0; r < 8; ++r) {
            int oc = nt * 16 + lm;
            float va = accA[nt][r];
            float vb = accB[nt][r];
            va = (va > 0.f) ? va : expm1f(va);   // elu, alpha=1
            vb = (vb > 0.f) ? vb : expm1f(vb);
            out[(size_t)(i0a + r + 8 * hl) * F_OUT + oc] = va;
            out[(size_t)(i0b + r + 8 * hl) * F_OUT + oc] = vb;
        }
    }
}

// ---------------------------------------------------------------------------
// Launch
// ---------------------------------------------------------------------------
extern "C" void kernel_launch(void* const* d_in, const int* in_sizes, int n_in,
                              void* d_out, int out_size, void* d_ws, size_t ws_size,
                              hipStream_t stream) {
    const float* x   = (const float*)d_in[0];   // 8192x256
    const int*   adj = (const int*)  d_in[1];   // 8192x8192
    const float* W   = (const float*)d_in[2];   // 256x128
    const float* a   = (const float*)d_in[3];   // 256x1
    float* out = (float*)d_out;                 // 8192x128

    char* ws = (char*)d_ws;
    float*  h    = (float*) (ws);                               // 4 MB
    __bf16* hp   = (__bf16*)(ws + 4194304);                     // 2 MB
    __bf16* Wp   = (__bf16*)(ws + 4194304 + 2097152);           // 64 KB
    float*  s1   = (float*) (ws + 6356992);                     // 32 KB
    float*  s2   = (float*) (ws + 6389760);                     // 32 KB
    float*  rmax = (float*) (ws + 6422528);                     // 32 KB
    float*  rinv = (float*) (ws + 6455296);                     // 32 KB

    pack_W_kernel       <<<128,  256, 0, stream>>>(W, Wp);
    gemm1_kernel        <<<128,  128, 0, stream>>>(x, Wp, h, hp);  // 512 tiles
    scores_kernel       <<<1024, 256, 0, stream>>>(h, a, s1, s2);  // 1 wave/row
    softmax_stats_kernel<<<1024, 256, 0, stream>>>(adj, s1, s2, rmax, rinv);
    gemm2_elu_kernel    <<<64,   128, 0, stream>>>(adj, s1, s2, rmax, rinv, hp, out);
}
